// Block_25391846654357
// MI455X (gfx1250) — matrix operand order repaired and checked by reading, not compiled, not run on hardware
//
#include <hip/hip_runtime.h>
#include <hip/hip_bf16.h>

// ---------------------------------------------------------------------------
// Types for CDNA5 WMMA (wave32): 16x16x32 f16 -> f32
// ---------------------------------------------------------------------------
typedef __attribute__((ext_vector_type(16))) _Float16 v16h;
typedef __attribute__((ext_vector_type(8)))  _Float16 v8h;
typedef __attribute__((ext_vector_type(4)))  _Float16 v4h;
typedef __attribute__((ext_vector_type(8)))  float    v8f;

#define TOK   32768   // B*T = 128*256
#define DIM   384
#define NHEAD 6
#define HD    64
#define TLEN  256

__device__ __forceinline__ v8f wmma_f16(v16h a, v16h b, v8f c) {
  return __builtin_amdgcn_wmma_f32_16x16x32_f16(false, a, false, b, (short)0, c,
                                                false, false);
}

// A-matrix fragment (16x32 f16, row-major source, ld in halves).
// Lane l: m = l&15, g = l>>4; VGPR v: K = 16*(v>>2) + 8*g + 2*(v&3) + h
// -> two contiguous 8-half runs at [m][k0+8g] and [m][k0+16+8g].
__device__ __forceinline__ v16h frag_ld_a(const _Float16* p, int ld) {
  int l = threadIdx.x & 31;
  const _Float16* q = p + (size_t)(l & 15) * ld + ((l >> 4) << 3);
  v8h lo = *(const v8h*)q;
  v8h hi = *(const v8h*)(q + 16);
  v16h r;
#pragma unroll
  for (int i = 0; i < 8; ++i) { r[i] = lo[i]; r[i + 8] = hi[i]; }
  return r;
}

// B-matrix fragment (32x16 f16) read from BT = [n][k] row-major (ld in halves).
// Lane l: n = l&15, g = l>>4; VGPR v: K = 16*(v>>2) + 8*g + 2*(v&3) + h
// -> two contiguous 8-half runs at [n][k0+8g] and [n][k0+16+8g].
__device__ __forceinline__ v16h frag_ld_b(const _Float16* p, int ld) {
  int l = threadIdx.x & 31;
  const _Float16* q = p + (size_t)(l & 15) * ld + ((l >> 4) << 3);
  v8h lo = *(const v8h*)q;
  v8h hi = *(const v8h*)(q + 16);
  v16h r;
#pragma unroll
  for (int i = 0; i < 8; ++i) { r[i] = lo[i]; r[i + 8] = hi[i]; }
  return r;
}

// CDNA5 async global->LDS copy, 16 bytes per active lane (ASYNCcnt-tracked).
// Generic LDS pointers keep the shared aperture in addr[63:32] only, so the
// low 32 bits are the LDS byte offset required by the VDST operand.
__device__ __forceinline__ void async_ld_lds_b128(void* lds, const void* gptr) {
  unsigned int loff = (unsigned int)(unsigned long long)lds;
  asm volatile("global_load_async_to_lds_b128 %0, %1, off"
               :: "v"(loff), "v"(gptr) : "memory");
}

__device__ __forceinline__ void wait_asynccnt0() {
  asm volatile("s_wait_asynccnt 0" ::: "memory");
}

// ---------------------------------------------------------------------------
// Weight prep: fp32 -> f16, transposed to BT[n][k]
// ---------------------------------------------------------------------------
__global__ void tcvt_kernel(const float* __restrict__ in, _Float16* __restrict__ out,
                            int K, int N) {
  long idx = (long)blockIdx.x * blockDim.x + threadIdx.x;
  if (idx >= (long)K * N) return;
  int n = (int)(idx / K), k = (int)(idx % K);
  out[idx] = (_Float16)in[(size_t)k * N + n];
}

// Combined QKV weight: out rows n in [0,1152): [q(0..383) | k | v],
// within a section: h = r/64, j = r%64; out[n*384 + d] = W*[h][d][j]
__global__ void qkvw_cvt_kernel(const float* __restrict__ Wq,
                                const float* __restrict__ Wk,
                                const float* __restrict__ Wv,
                                _Float16* __restrict__ out) {
  int idx = blockIdx.x * blockDim.x + threadIdx.x;
  if (idx >= 1152 * 384) return;
  int n = idx / 384, d = idx % 384;
  int sec = n / 384, r = n % 384;
  int h = r / 64, j = r % 64;
  const float* W = (sec == 0) ? Wq : (sec == 1 ? Wk : Wv);
  out[idx] = (_Float16)W[((size_t)h * 384 + d) * 64 + j];
}

// ---------------------------------------------------------------------------
// LayerNorm: one wave per 384-wide row, f32 in -> f16 out
// ---------------------------------------------------------------------------
__global__ __launch_bounds__(256) void ln_kernel(const float* __restrict__ x,
                                                 const float* __restrict__ gam,
                                                 const float* __restrict__ bet,
                                                 _Float16* __restrict__ out) {
  int row = blockIdx.x * 8 + (threadIdx.x >> 5);
  int lane = threadIdx.x & 31;
  const float* xr = x + (size_t)row * DIM;
  float v[12], s = 0.f, s2 = 0.f;
#pragma unroll
  for (int i = 0; i < 12; ++i) {
    v[i] = xr[lane + 32 * i];
    s += v[i];
    s2 += v[i] * v[i];
  }
#pragma unroll
  for (int m = 16; m >= 1; m >>= 1) {
    s  += __shfl_xor(s,  m, 32);
    s2 += __shfl_xor(s2, m, 32);
  }
  float mean = s * (1.f / DIM);
  float var  = s2 * (1.f / DIM) - mean * mean;
  float rstd = rsqrtf(var + 1e-5f);
#pragma unroll
  for (int i = 0; i < 12; ++i) {
    int d = lane + 32 * i;
    out[(size_t)row * DIM + d] = (_Float16)((v[i] - mean) * rstd * gam[d] + bet[d]);
  }
}

// ---------------------------------------------------------------------------
// WMMA GEMM: C[M,N] = A[M,K] * BT[N,K]^T  (+bias)(+resid)(ReLU?), out f16/f32
// Workgroup = 256 thr = 8 waves -> 128x128 tile; wave -> 32x64:
// 2 A-fragments, 4 B-fragments (shared across both row tiles), 8 WMMA/k-step.
// ---------------------------------------------------------------------------
template <int OUTF16, int RELU>
__global__ __launch_bounds__(256) void gemm_wmma_kernel(
    const _Float16* __restrict__ A, const _Float16* __restrict__ BT,
    const float* __restrict__ bias, const float* __restrict__ resid,
    void* __restrict__ Cout, int M, int N, int K) {
  int tiles_n = N >> 7;
  int bm = blockIdx.x / tiles_n, bn = blockIdx.x % tiles_n;
  int wave = threadIdx.x >> 5;
  int m0 = bm * 128 + (wave >> 1) * 32;
  int n0 = bn * 128 + (wave & 1) * 64;
  (void)M;

  const v8f vz = {};
  v8f acc[2][4];
#pragma unroll
  for (int mt = 0; mt < 2; ++mt)
#pragma unroll
    for (int j = 0; j < 4; ++j) acc[mt][j] = vz;

  const _Float16* A0 = A + (size_t)m0 * K;
  const _Float16* A1 = A0 + (size_t)16 * K;
  for (int k0 = 0; k0 < K; k0 += 32) {
    __builtin_prefetch(A0 + k0 + 128, 0, 1);
    __builtin_prefetch(A1 + k0 + 128, 0, 1);
    v16h a0 = frag_ld_a(A0 + k0, K);
    v16h a1 = frag_ld_a(A1 + k0, K);
#pragma unroll
    for (int j = 0; j < 4; ++j) {
      v16h bf = frag_ld_b(BT + (size_t)(n0 + 16 * j) * K + k0, K);
      acc[0][j] = wmma_f16(a0, bf, acc[0][j]);
      acc[1][j] = wmma_f16(a1, bf, acc[1][j]);
    }
  }

  int l = threadIdx.x & 31, col = l & 15, g = l >> 4;
#pragma unroll
  for (int j = 0; j < 4; ++j) {
    int n = n0 + 16 * j + col;
    float bv = bias ? bias[n] : 0.f;
#pragma unroll
    for (int mt = 0; mt < 2; ++mt) {
#pragma unroll
      for (int r = 0; r < 8; ++r) {
        int m = m0 + 16 * mt + r + 8 * g;
        float val = acc[mt][j][r] + bv;
        if (resid) val += resid[(size_t)m * N + n];
        if (RELU) val = fmaxf(val, 0.f);
        if (OUTF16) ((_Float16*)Cout)[(size_t)m * N + n] = (_Float16)val;
        else        ((float*)Cout)[(size_t)m * N + n] = val;
      }
    }
  }
}

// ---------------------------------------------------------------------------
// Flash attention: one block (512 thr = 16 waves) per (b,h).
// qkv f16: [token, 1152] = [q | k | v], each section [h*64 + j].
// Wave w owns query rows t0 = 16w. Keys processed in chunks of 32 (causal).
// K and V chunks staged via global_load_async_to_lds_b128 (ASYNCcnt).
// ---------------------------------------------------------------------------
__global__ __launch_bounds__(512) void attn_kernel(const _Float16* __restrict__ qkv,
                                                   _Float16* __restrict__ att) {
  __shared__ _Float16 sK [32][72];     // [s_local][hd]        (B^T for scores)
  __shared__ _Float16 sVr[32][72];     // [s_local][hd]        (V rows, async)
  __shared__ _Float16 sVt[64][40];     // [hd][s_local]        (B^T for P*V)
  __shared__ _Float16 sP [16][16][40]; // [wave][row][s_local] (A for P*V)

  int b = blockIdx.x / NHEAD, h = blockIdx.x % NHEAD;
  int tid = threadIdx.x;
  int w = tid >> 5, lane = tid & 31;
  int col = lane & 15, g = lane >> 4;
  int t0 = w * 16;

  const size_t ld = 1152;
  const _Float16* qbase = qkv + (size_t)(b * TLEN) * ld + h * HD;
  const _Float16* kbase = qbase + 384;
  const _Float16* vbase = qbase + 768;

  // Q fragments are loop-invariant (K=64 -> two 16x32 A-fragments)
  v16h qf0 = frag_ld_a(qbase + (size_t)t0 * ld + 0,  (int)ld);
  v16h qf1 = frag_ld_a(qbase + (size_t)t0 * ld + 32, (int)ld);

  const v8f vz = {};
  v8f oacc[4];
#pragma unroll
  for (int j = 0; j < 4; ++j) oacc[j] = vz;
  float m_run[8], l_run[8];
#pragma unroll
  for (int r = 0; r < 8; ++r) { m_run[r] = -INFINITY; l_run[r] = 0.f; }

  const float inv_scale = 0.05103103630798287f;  // 1/sqrt(384) (matches ref!)

  for (int c = 0; c < 8; ++c) {
    int s0 = c * 32;
    __syncthreads();  // previous chunk's compute done; sK/sVr/sVt reusable
    // --- async-stage K and V chunks: 32 rows x 64 halves = 256 x 16B each ---
    if (tid < 256) {
      int rr  = tid >> 3;         // 0..31
      int seg = (tid & 7) * 8;    // 0,8,...,56 (halves)
      async_ld_lds_b128(&sK [rr][seg], kbase + (size_t)(s0 + rr) * ld + seg);
      async_ld_lds_b128(&sVr[rr][seg], vbase + (size_t)(s0 + rr) * ld + seg);
    }
    wait_asynccnt0();
    __syncthreads();  // all async data landed in LDS
    // --- transpose V within LDS: sVt[hd][s] = sVr[s][hd] ---
#pragma unroll
    for (int i = 0; i < 4; ++i) {
      int e = tid * 4 + i;        // 0..2047
      int s = e >> 6, j = e & 63;
      sVt[j][s] = sVr[s][j];
    }
    __syncthreads();
    if (t0 + 15 < s0) continue;   // fully masked for this wave

    // --- scores: S = Q K^T (two 16-col groups, K=64 over two steps) ---
    v8f sc0 = vz, sc1 = vz;
    sc0 = wmma_f16(qf0, frag_ld_b(&sK[0][0],  72), sc0);
    sc0 = wmma_f16(qf1, frag_ld_b(&sK[0][32], 72), sc0);
    sc1 = wmma_f16(qf0, frag_ld_b(&sK[16][0],  72), sc1);
    sc1 = wmma_f16(qf1, frag_ld_b(&sK[16][32], 72), sc1);

    // --- online softmax (row = r + 8g, col = l&15) ---
#pragma unroll
    for (int r = 0; r < 8; ++r) {
      int t = t0 + r + 8 * g;
      float a0 = sc0[r] * inv_scale;
      float a1 = sc1[r] * inv_scale;
      if (s0 + col > t)      a0 = -INFINITY;
      if (s0 + 16 + col > t) a1 = -INFINITY;
      float cm = fmaxf(a0, a1);
#pragma unroll
      for (int m = 8; m >= 1; m >>= 1) cm = fmaxf(cm, __shfl_xor(cm, m, 32));
      float nm = fmaxf(m_run[r], cm);
      float scale = __expf(m_run[r] - nm);
      m_run[r] = nm;
      float p0 = __expf(a0 - nm);
      float p1 = __expf(a1 - nm);
      float rs = p0 + p1;
#pragma unroll
      for (int m = 8; m >= 1; m >>= 1) rs += __shfl_xor(rs, m, 32);
      l_run[r] = l_run[r] * scale + rs;
#pragma unroll
      for (int og = 0; og < 4; ++og) oacc[og][r] *= scale;
      sP[w][r + 8 * g][col]      = (_Float16)p0;
      sP[w][r + 8 * g][16 + col] = (_Float16)p1;
    }

    // intra-wave LDS store->load ordering for P re-layout
    asm volatile("s_wait_dscnt 0" ::: "memory");

    // --- O += P * V  (A = P 16x32 from LDS, B^T = sVt) ---
    v16h pf = frag_ld_a(&sP[w][0][0], 40);
#pragma unroll
    for (int og = 0; og < 4; ++og)
      oacc[og] = wmma_f16(pf, frag_ld_b(&sVt[16 * og][0], 40), oacc[og]);
  }

  // --- normalize and store concat-head output, f16 ---
#pragma unroll
  for (int og = 0; og < 4; ++og) {
#pragma unroll
    for (int r = 0; r < 8; ++r) {
      int t = t0 + r + 8 * g;
      float val = oacc[og][r] / l_run[r];
      att[((size_t)(b * TLEN + t)) * DIM + h * HD + 16 * og + col] = (_Float16)val;
    }
  }
}

// ---------------------------------------------------------------------------
// Orchestration
// ---------------------------------------------------------------------------
extern "C" void kernel_launch(void* const* d_in, const int* in_sizes, int n_in,
                              void* d_out, int out_size, void* d_ws, size_t ws_size,
                              hipStream_t stream) {
  (void)in_sizes; (void)n_in; (void)out_size; (void)ws_size;
  const float* x   = (const float*)d_in[0];
  const float* Wq  = (const float*)d_in[1];
  const float* Wk  = (const float*)d_in[2];
  const float* Wv  = (const float*)d_in[3];
  const float* Wp  = (const float*)d_in[4];
  const float* bp  = (const float*)d_in[5];
  const float* W1  = (const float*)d_in[6];
  const float* b1  = (const float*)d_in[7];
  const float* W2  = (const float*)d_in[8];
  const float* b2  = (const float*)d_in[9];
  const float* g1  = (const float*)d_in[10];
  const float* be1 = (const float*)d_in[11];
  const float* g2  = (const float*)d_in[12];
  const float* be2 = (const float*)d_in[13];

  char* ws = (char*)d_ws;
  size_t off = 0;
  auto alloc = [&](size_t bytes) {
    off = (off + 255) & ~(size_t)255;
    void* p = ws + off;
    off += bytes;
    return p;
  };

  _Float16* h1    = (_Float16*)alloc((size_t)TOK * DIM * 2);
  _Float16* WqkvT = (_Float16*)alloc((size_t)1152 * 384 * 2);
  _Float16* qkv   = (_Float16*)alloc((size_t)TOK * 1152 * 2);
  _Float16* att   = (_Float16*)alloc((size_t)TOK * DIM * 2);
  _Float16* WpT   = (_Float16*)alloc((size_t)384 * 384 * 2);
  float*    x1    = (float*)   alloc((size_t)TOK * DIM * 4);
  _Float16* h2    = (_Float16*)alloc((size_t)TOK * DIM * 2);
  _Float16* W1T   = (_Float16*)alloc((size_t)1536 * 384 * 2);
  _Float16* a1    = (_Float16*)alloc((size_t)TOK * 1536 * 2);
  _Float16* W2T   = (_Float16*)alloc((size_t)384 * 1536 * 2);

  // 1) weight conversion (f32 -> f16, transposed)
  qkvw_cvt_kernel<<<(1152 * 384 + 255) / 256, 256, 0, stream>>>(Wq, Wk, Wv, WqkvT);
  tcvt_kernel<<<(384 * 384 + 255) / 256, 256, 0, stream>>>(Wp, WpT, 384, 384);
  tcvt_kernel<<<(384 * 1536 + 255) / 256, 256, 0, stream>>>(W1, W1T, 384, 1536);
  tcvt_kernel<<<(1536 * 384 + 255) / 256, 256, 0, stream>>>(W2, W2T, 1536, 384);

  // 2) LN1
  ln_kernel<<<TOK / 8, 256, 0, stream>>>(x, g1, be1, h1);

  // 3) QKV: [32768,384] x [384,1152] -> f16
  gemm_wmma_kernel<1, 0><<<(TOK / 128) * (1152 / 128), 256, 0, stream>>>(
      h1, WqkvT, nullptr, nullptr, qkv, TOK, 1152, DIM);

  // 4) causal flash attention -> att f16 (heads concatenated)
  attn_kernel<<<128 * NHEAD, 512, 0, stream>>>(qkv, att);

  // 5) proj + bias + residual(x) -> x1 f32
  gemm_wmma_kernel<0, 0><<<(TOK / 128) * (DIM / 128), 256, 0, stream>>>(
      att, WpT, bp, x, x1, TOK, DIM, DIM);

  // 6) LN2
  ln_kernel<<<TOK / 8, 256, 0, stream>>>(x1, g2, be2, h2);

  // 7) FFN1 + bias + ReLU -> a1 f16
  gemm_wmma_kernel<1, 1><<<(TOK / 128) * (1536 / 128), 256, 0, stream>>>(
      h2, W1T, b1, nullptr, a1, TOK, 1536, DIM);

  // 8) FFN2 + bias + residual(x1) -> d_out f32
  gemm_wmma_kernel<0, 0><<<(TOK / 128) * (DIM / 128), 256, 0, stream>>>(
      a1, W2T, b2, x1, d_out, TOK, DIM, 1536);
}
